// TransferNet_7584912244791
// MI455X (gfx1250) — compile-verified, matrix operand order checked by reading, not run on hardware
//
#include <hip/hip_runtime.h>
#include <hip/hip_bf16.h>

typedef unsigned short u16;
typedef __attribute__((ext_vector_type(16))) __bf16 v16bf;
typedef __attribute__((ext_vector_type(8)))  float  v8f;
typedef int vint4 __attribute__((vector_size(16)));   // matches builtin's V4i param

static constexpr int D     = 768;
static constexpr int D2    = 1536;
static constexpr int D4    = 3072;
static constexpr int LPARA = 512;
static constexpr int LQ    = 32;
static constexpr int NENT  = 10000;
static constexpr int MME   = 30000;
static constexpr int NPAIR = 20000;
static constexpr int NTOT  = NENT + 2 * NPAIR;   // 50000
static constexpr int SFEAT = 256;

__device__ __forceinline__ u16 f2bf(float f) {
    unsigned u = __float_as_uint(f);
    unsigned r = u + 0x7FFFu + ((u >> 16) & 1u);
    return (u16)(r >> 16);
}
__device__ __forceinline__ float gelu_f(float x) {
    return 0.5f * x * (1.0f + erff(x * 0.70710678118654752f));
}

// ---------------- CDNA5 async global->LDS copy (guarded; sync fallback) ------
#if __has_builtin(__builtin_amdgcn_global_load_async_to_lds_b128) && \
    __has_builtin(__builtin_amdgcn_s_wait_asynccnt)
#define HAVE_ASYNC_LDS 1
#define CP16(lds, glb)                                                         \
    __builtin_amdgcn_global_load_async_to_lds_b128(                            \
        (__attribute__((address_space(1))) vint4*)(__UINTPTR_TYPE__)(glb),     \
        (__attribute__((address_space(3))) vint4*)(__UINTPTR_TYPE__)(lds),     \
        0, 0)
#define ASYNC_WAIT(n) __builtin_amdgcn_s_wait_asynccnt(n)
#else
#define HAVE_ASYNC_LDS 0
#if defined(__gfx1250__)
#warning "gfx1250 async-to-LDS builtins unavailable; using synchronous LDS staging"
#endif
#define CP16(lds, glb) (*(uint4*)(lds) = *(const uint4*)(glb))
#define ASYNC_WAIT(n) ((void)0)
#endif

// ---------------------------------------------------------------- cvt f32->bf16
__global__ void k_cvt_bf16(const float* __restrict__ s, u16* __restrict__ d, int n) {
    int i = blockIdx.x * blockDim.x + threadIdx.x;
    if (i < n) d[i] = f2bf(s[i]);
}

// cvt + transpose: s[K][N] f32 row-major  ->  d[N][K] bf16 row-major
__global__ void k_cvt_bf16_T(const float* __restrict__ s, u16* __restrict__ d,
                             int Kdim, int Ndim) {
    int idx = blockIdx.x * blockDim.x + threadIdx.x;
    if (idx >= Kdim * Ndim) return;
    int n = idx / Kdim, k = idx - n * Kdim;
    d[idx] = f2bf(s[(size_t)k * Ndim + n]);
}

// ------------------------------------------------- mention gather + segment sum
__global__ void k_seg_accum(const float* __restrict__ para, const int* __restrict__ pos,
                            const int* __restrict__ mid, float* __restrict__ ent_sum,
                            float* __restrict__ cnt) {
    int m = blockIdx.x;
    if (m >= MME) return;
    int pid = pos[m * 3 + 0], tl = pos[m * 3 + 1], tr = pos[m * 3 + 2];
    int id  = mid[m];
    const float* L = para + ((size_t)pid * LPARA + tl) * D;
    const float* R = para + ((size_t)pid * LPARA + tr) * D;
    float* dst = ent_sum + (size_t)id * D2;
    for (int j = threadIdx.x; j < D; j += blockDim.x) {
        atomicAdd(dst + j, L[j]);
        atomicAdd(dst + D + j, R[j]);
    }
    if (threadIdx.x == 0) atomicAdd(cnt + id, 1.0f);
}

// ---------------------------- build X = [ent_mean ; sub_cat ; obj_cat] in bf16
__global__ void k_build_X(const float* __restrict__ para, const float* __restrict__ ent_sum,
                          const float* __restrict__ cnt, const int* __restrict__ ppos,
                          u16* __restrict__ X) {
    int r = blockIdx.x;             // 0..NTOT-1
    u16* xr = X + (size_t)r * D2;
    if (r < NENT) {
        float inv = 1.0f / cnt[r];
        const float* s = ent_sum + (size_t)r * D2;
        for (int j = threadIdx.x; j < D2; j += blockDim.x) xr[j] = f2bf(s[j] * inv);
    } else {
        int p, c1, c2;
        if (r < NENT + NPAIR) { p = r - NENT; c1 = 1; c2 = 2; }
        else                  { p = r - NENT - NPAIR; c1 = 3; c2 = 4; }
        int pid = ppos[p * 5 + 0];
        const float* L = para + ((size_t)pid * LPARA + ppos[p * 5 + c1]) * D;
        const float* R = para + ((size_t)pid * LPARA + ppos[p * 5 + c2]) * D;
        for (int j = threadIdx.x; j < D; j += blockDim.x) {
            xr[j]     = f2bf(L[j]);
            xr[D + j] = f2bf(R[j]);
        }
    }
}

// -------------------------------- pair_inp = [sub, obj, obj-sub, sub*obj] bf16
__global__ void k_build_pair(const float* __restrict__ F, u16* __restrict__ P) {
    int p = blockIdx.x;             // 0..NPAIR-1
    const float* s = F + (size_t)(NENT + p) * D;
    const float* o = F + (size_t)(NENT + NPAIR + p) * D;
    u16* out = P + (size_t)p * D4;
    for (int j = threadIdx.x; j < D; j += blockDim.x) {
        float sv = s[j], ov = o[j];
        out[j]         = f2bf(sv);
        out[D + j]     = f2bf(ov);
        out[2 * D + j] = f2bf(ov - sv);
        out[3 * D + j] = f2bf(sv * ov);
    }
}

// ---------------------------------------- ctx_h per step + hop attention (tiny)
__global__ void k_ctx(const float* __restrict__ q, const float* __restrict__ qw,
                      const float* __restrict__ stW1, const float* __restrict__ stb1,
                      const float* __restrict__ stW2, const float* __restrict__ stb2,
                      const float* __restrict__ hW1, const float* __restrict__ hb1,
                      const float* __restrict__ hW2, const float* __restrict__ hb2,
                      float* __restrict__ ctx, float* __restrict__ hop) {
    __shared__ float h1[D], cq[D], red[256], dist[LQ], hh[SFEAT];
    int tid = threadIdx.x;
    for (int t = 0; t < 3; ++t) {
        const float* W1 = stW1 + (size_t)t * D * D;
        const float* b1 = stb1 + t * D;
        const float* W2 = stW2 + (size_t)t * D * D;
        const float* b2 = stb2 + t * D;
        for (int o = tid; o < D; o += blockDim.x) {
            float a = b1[o];
            for (int i = 0; i < D; ++i) a += q[i] * W1[(size_t)i * D + o];
            h1[o] = gelu_f(a);
        }
        __syncthreads();
        for (int o = tid; o < D; o += blockDim.x) {
            float a = b2[o];
            for (int i = 0; i < D; ++i) a += h1[i] * W2[(size_t)i * D + o];
            cq[o] = a;
        }
        __syncthreads();
        for (int l = 0; l < LQ; ++l) {
            float part = 0.f;
            for (int i = tid; i < D; i += blockDim.x) part += cq[i] * qw[(size_t)l * D + i];
            red[tid] = part; __syncthreads();
            for (int s = 128; s > 0; s >>= 1) { if (tid < s) red[tid] += red[tid + s]; __syncthreads(); }
            if (tid == 0) dist[l] = red[0];
            __syncthreads();
        }
        if (tid == 0) {
            float mx = dist[0];
            for (int l = 1; l < LQ; ++l) mx = fmaxf(mx, dist[l]);
            float sum = 0.f;
            for (int l = 0; l < LQ; ++l) { dist[l] = expf(dist[l] - mx); sum += dist[l]; }
            for (int l = 0; l < LQ; ++l) dist[l] /= sum;
        }
        __syncthreads();
        for (int o = tid; o < D; o += blockDim.x) {
            float a = cq[o];
            for (int l = 0; l < LQ; ++l) a += dist[l] * qw[(size_t)l * D + o];
            ctx[(size_t)t * D + o] = a;
        }
        __syncthreads();
    }
    for (int o = tid; o < SFEAT; o += blockDim.x) {
        float a = hb1[o];
        for (int i = 0; i < D; ++i) a += q[i] * hW1[(size_t)i * SFEAT + o];
        hh[o] = gelu_f(a);
    }
    __syncthreads();
    if (tid == 0) {
        float lg[3];
        for (int t = 0; t < 3; ++t) {
            float a = hb2[t];
            for (int i = 0; i < SFEAT; ++i) a += hh[i] * hW2[i * 3 + t];
            lg[t] = a;
        }
        float mx = fmaxf(lg[0], fmaxf(lg[1], lg[2]));
        float s = 0.f;
        for (int t = 0; t < 3; ++t) { lg[t] = expf(lg[t] - mx); s += lg[t]; }
        for (int t = 0; t < 3; ++t) hop[t] = lg[t] / s;
    }
}

// ----------------------- score-MLP input: feat * ctx_t per batched row (bf16)
__global__ void k_build_sin(const float* __restrict__ F, const float* __restrict__ PE,
                            const float* __restrict__ ctx, u16* __restrict__ S) {
    int r = blockIdx.x;             // 0..NTOT-1
    u16* out = S + (size_t)r * D;
    const float* x; const float* c;
    if (r < NENT)            { x = F  + (size_t)r * D;                  c = ctx;         }
    else if (r < NENT+NPAIR) { x = PE + (size_t)(r - NENT) * D;         c = ctx + D;     }
    else                     { x = PE + (size_t)(r - NENT - NPAIR) * D; c = ctx + 2 * D; }
    for (int j = threadIdx.x; j < D; j += blockDim.x) out[j] = f2bf(x[j] * c[j]);
}

// ------------------------------------------------ WMMA bf16 GEMM, 128x128 tiles
// C[M,N] = epi(A[M,K](bf16) @ Bt[N,K](bf16, pre-transposed weights) + bias[N])
// Double-buffered LDS staging via GLOBAL_LOAD_ASYNC_TO_LDS_B128 + s_wait_asynccnt.
static constexpr int BM = 128, BN = 128, BK = 32, LDSS = 40;
struct Q2 { uint4 a; uint4 b; };

__global__ __launch_bounds__(256) void k_gemm_wmma(
    const u16* __restrict__ A, const u16* __restrict__ Bt,
    const float* __restrict__ bias, void* __restrict__ Cout,
    int M, int N, int K, int doGelu, int outBf16) {
    __shared__ alignas(16) u16 As[2][BM * LDSS];
    __shared__ alignas(16) u16 Bs[2][BN * LDSS];   // [n][k]

    const int tid  = threadIdx.x;
    const int lane = tid & 31;
    const int wave = tid >> 5;
    const int wm   = wave >> 1;          // 0..3 -> 32-row slab
    const int wn   = wave & 1;           // 0..1 -> 64-col slab
    const int half = lane >> 4;
    const int l16  = lane & 15;
    const int rowBase = blockIdx.y * BM;
    const int colBase = blockIdx.x * BN;

    const int ldr = tid >> 2;            // 0..63
    const int ldc = (tid & 3) * 8;       // 0,8,16,24

    // Clamp A rows: keeps the per-wave async-op count uniform; clamped rows only
    // feed C rows with gm>=M which the epilogue masks out.
    int ar0 = rowBase + ldr;      if (ar0 >= M) ar0 = M - 1;
    int ar1 = rowBase + ldr + 64; if (ar1 >= M) ar1 = M - 1;
    const int bn0 = colBase + ldr;       // N is a multiple of BN for all GEMMs here
    const int bn1 = colBase + ldr + 64;

    v8f zero = {0.f, 0.f, 0.f, 0.f, 0.f, 0.f, 0.f, 0.f};
    v8f acc[2][4];
    #pragma unroll
    for (int mf = 0; mf < 2; ++mf)
        #pragma unroll
        for (int nf = 0; nf < 4; ++nf) acc[mf][nf] = zero;

    auto issue = [&](int k0, int buf) {
        CP16(&As[buf][ldr * LDSS + ldc],        A  + (size_t)ar0 * K + k0 + ldc);
        CP16(&As[buf][(ldr + 64) * LDSS + ldc], A  + (size_t)ar1 * K + k0 + ldc);
        CP16(&Bs[buf][ldr * LDSS + ldc],        Bt + (size_t)bn0 * K + k0 + ldc);
        CP16(&Bs[buf][(ldr + 64) * LDSS + ldc], Bt + (size_t)bn1 * K + k0 + ldc);
    };

    issue(0, 0);
    int buf = 0;
    for (int k0 = 0; k0 < K; k0 += BK, buf ^= 1) {
        const bool hasNext = (k0 + BK < K);
        if (hasNext) {
            issue(k0 + BK, buf ^ 1);     // prefetch next tile into other buffer
            ASYNC_WAIT(4);               // retire this tile's 4 copies (in-order)
        } else {
            ASYNC_WAIT(0);
        }
        __syncthreads();

        const u16* Ab = &As[buf][0];
        const u16* Bb = &Bs[buf][0];
        v16bf afr[2], bfr[4];
        #pragma unroll
        for (int mf = 0; mf < 2; ++mf) {
            int ar = wm * 32 + mf * 16 + l16;
            Q2 qd;
            qd.a = *(const uint4*)(Ab + ar * LDSS + half * 8);
            qd.b = *(const uint4*)(Ab + ar * LDSS + 16 + half * 8);
            afr[mf] = __builtin_bit_cast(v16bf, qd);
        }
        #pragma unroll
        for (int nf = 0; nf < 4; ++nf) {
            int bn = wn * 64 + nf * 16 + l16;
            Q2 qd;
            qd.a = *(const uint4*)(Bb + bn * LDSS + half * 8);
            qd.b = *(const uint4*)(Bb + bn * LDSS + 16 + half * 8);
            bfr[nf] = __builtin_bit_cast(v16bf, qd);
        }
        #pragma unroll
        for (int mf = 0; mf < 2; ++mf)
            #pragma unroll
            for (int nf = 0; nf < 4; ++nf)
                acc[mf][nf] = __builtin_amdgcn_wmma_f32_16x16x32_bf16(
                    false, afr[mf], false, bfr[nf], (short)0, acc[mf][nf], false, false);
        __syncthreads();
    }

    // epilogue: C/D layout — lane l16 = column, VGPR r -> row r (+8 for lanes>=16)
    #pragma unroll
    for (int mf = 0; mf < 2; ++mf) {
        #pragma unroll
        for (int nf = 0; nf < 4; ++nf) {
            int gn = colBase + wn * 64 + nf * 16 + l16;
            float bv = bias[gn];
            #pragma unroll
            for (int r = 0; r < 8; ++r) {
                int gm = rowBase + wm * 32 + mf * 16 + half * 8 + r;
                if (gm >= M) continue;
                float v = acc[mf][nf][r] + bv;
                if (doGelu) v = gelu_f(v);
                if (outBf16) ((u16*)Cout)[(size_t)gm * N + gn] = f2bf(v);
                else         ((float*)Cout)[(size_t)gm * N + gn] = v;
            }
        }
    }
}

// -------------------------------------------- per-row dot with sW2 (+sb2) -> score
__global__ void k_score(const float* __restrict__ HS, const float* __restrict__ w,
                        const float* __restrict__ b, float* __restrict__ scores, int n) {
    int row  = blockIdx.x * 8 + (threadIdx.x >> 5);
    int lane = threadIdx.x & 31;
    if (row >= n) return;
    const float* h = HS + (size_t)row * SFEAT;
    float acc = 0.f;
    for (int j = lane; j < SFEAT; j += 32) acc += h[j] * w[j];
    for (int off = 16; off > 0; off >>= 1) acc += __shfl_xor(acc, off, 32);
    if (lane == 0) scores[row] = acc + b[0];
}

// ----------------------------------------- step 0: softmax over entities + clamp
__global__ void k_step0(const float* __restrict__ scores, float* __restrict__ probs) {
    __shared__ float red[256];
    int tid = threadIdx.x;
    float mx = -1e30f;
    for (int i = tid; i < NENT; i += blockDim.x) mx = fmaxf(mx, scores[i]);
    red[tid] = mx; __syncthreads();
    for (int s = 128; s > 0; s >>= 1) { if (tid < s) red[tid] = fmaxf(red[tid], red[tid + s]); __syncthreads(); }
    mx = red[0]; __syncthreads();
    float sum = 0.f;
    for (int i = tid; i < NENT; i += blockDim.x) sum += expf(scores[i] - mx);
    red[tid] = sum; __syncthreads();
    for (int s = 128; s > 0; s >>= 1) { if (tid < s) red[tid] += red[tid + s]; __syncthreads(); }
    sum = red[0];
    for (int i = tid; i < NENT; i += blockDim.x) {
        float v = expf(scores[i] - mx) / sum;
        probs[i] = v / fmaxf(v, 1.0f);
    }
}

// --------------------- step t>0: sigmoid pair prob, scatter-add to next entity dist
__global__ void k_pair_step(const float* __restrict__ sc, const float* __restrict__ prev,
                            const int* __restrict__ so, float* __restrict__ acc) {
    int p = blockIdx.x * blockDim.x + threadIdx.x;
    if (p >= NPAIR) return;
    float pr = 1.0f / (1.0f + expf(-sc[p]));
    atomicAdd(acc + so[p * 2 + 1], prev[so[p * 2 + 0]] * pr);
}

__global__ void k_norm(const float* __restrict__ acc, float* __restrict__ probs_t) {
    int i = blockIdx.x * blockDim.x + threadIdx.x;
    if (i >= NENT) return;
    float v = acc[i];
    probs_t[i] = v / fmaxf(v, 1.0f);
}

__global__ void k_final(const float* __restrict__ probs, const float* __restrict__ hop,
                        float* __restrict__ out) {
    int i = blockIdx.x * blockDim.x + threadIdx.x;
    if (i >= NENT) return;
    out[i] = hop[0] * probs[i] + hop[1] * probs[NENT + i] + hop[2] * probs[2 * NENT + i];
}

// =============================================================================
extern "C" void kernel_launch(void* const* d_in, const int* in_sizes, int n_in,
                              void* d_out, int out_size, void* d_ws, size_t ws_size,
                              hipStream_t stream) {
    (void)in_sizes; (void)n_in; (void)out_size; (void)ws_size;

    const float* para = (const float*)d_in[0];
    const float* q    = (const float*)d_in[1];
    const float* qw   = (const float*)d_in[2];
    const float* stW1 = (const float*)d_in[3];
    const float* stb1 = (const float*)d_in[4];
    const float* stW2 = (const float*)d_in[5];
    const float* stb2 = (const float*)d_in[6];
    const float* mW1  = (const float*)d_in[7];
    const float* mb1  = (const float*)d_in[8];
    const float* mW2  = (const float*)d_in[9];
    const float* mb2  = (const float*)d_in[10];
    const float* pW1  = (const float*)d_in[11];
    const float* pb1  = (const float*)d_in[12];
    const float* pW2  = (const float*)d_in[13];
    const float* pb2  = (const float*)d_in[14];
    const float* sW1  = (const float*)d_in[15];
    const float* sb1  = (const float*)d_in[16];
    const float* sW2  = (const float*)d_in[17];
    const float* sb2  = (const float*)d_in[18];
    const float* hW1  = (const float*)d_in[19];
    const float* hb1  = (const float*)d_in[20];
    const float* hW2  = (const float*)d_in[21];
    const float* hb2  = (const float*)d_in[22];
    const int*   mpos = (const int*)d_in[23];
    const int*   mid  = (const int*)d_in[24];
    const int*   ppos = (const int*)d_in[25];
    const int*   pso  = (const int*)d_in[26];

    // ----- workspace carve (lifetimes allow arena reuse) -----
    char*  w   = (char*)d_ws;
    size_t off = 0;
    auto carve = [&](size_t bytes) -> void* {
        void* p = w + off;
        off += (bytes + 255) & ~(size_t)255;
        return p;
    };
    u16* mW1b = (u16*)carve((size_t)D2 * D * 2);       // [768][1536]
    u16* mW2b = (u16*)carve((size_t)D  * D * 2);       // [768][768]
    u16* pW1b = (u16*)carve((size_t)D4 * D * 2);       // [768][3072]
    u16* pW2b = (u16*)carve((size_t)D  * D * 2);       // [768][768]
    u16* sW1b = (u16*)carve((size_t)D  * SFEAT * 2);   // [256][768]
    float* ctx    = (float*)carve(3 * D * 4);
    float* hop    = (float*)carve(3 * 4);
    float* scores = (float*)carve((size_t)NTOT * 4);
    float* probs  = (float*)carve((size_t)3 * NENT * 4);
    float* cnt    = (float*)carve((size_t)NENT * 4);
    float* e1     = (float*)carve((size_t)NENT * 4);
    float* e2     = (float*)carve((size_t)NENT * 4);
    void* arenaA = carve((size_t)NENT * D2 * 4);        // ent_sum f32  -> pair_emb f32
    void* arenaX = carve((size_t)NTOT * D2 * 2);        // X bf16 -> pair_inp bf16 -> SIN bf16
    void* arenaH = carve((size_t)NTOT * D  * 2);        // H1 bf16 -> HP bf16 -> HS f32
    void* arenaF = carve((size_t)NTOT * D  * 4);        // F f32 (ent_emb|sub_feat|obj_feat)

    float* ent_sum = (float*)arenaA;
    float* PE      = (float*)arenaA;
    u16*   X       = (u16*)arenaX;
    u16*   PIN     = (u16*)arenaX;
    u16*   SIN     = (u16*)arenaX;
    u16*   H1      = (u16*)arenaH;
    u16*   HP      = (u16*)arenaH;
    float* HS      = (float*)arenaH;
    float* F       = (float*)arenaF;
    float* out     = (float*)d_out;

    // ----- zero accumulators (every call; graph replays re-zero) -----
    (void)hipMemsetAsync(ent_sum, 0, (size_t)NENT * D2 * 4, stream);
    (void)hipMemsetAsync(cnt, 0, (size_t)NENT * 4, stream);
    (void)hipMemsetAsync(e1, 0, (size_t)NENT * 4, stream);
    (void)hipMemsetAsync(e2, 0, (size_t)NENT * 4, stream);

    // ----- weights -> bf16, transposed to [N][K] for the GEMM B operand -----
    auto cvtT = [&](const float* s, u16* d, int Kd, int Nd) {
        int n = Kd * Nd;
        k_cvt_bf16_T<<<(n + 255) / 256, 256, 0, stream>>>(s, d, Kd, Nd);
    };
    cvtT(mW1, mW1b, D2, D);
    cvtT(mW2, mW2b, D,  D);
    cvtT(pW1, pW1b, D4, D);
    cvtT(pW2, pW2b, D,  D);
    cvtT(sW1, sW1b, D,  SFEAT);

    // ----- tiny sequential context + hop attention -----
    k_ctx<<<1, 256, 0, stream>>>(q, qw, stW1, stb1, stW2, stb2, hW1, hb1, hW2, hb2, ctx, hop);

    // ----- entity mention segment-sum, build batched X -----
    k_seg_accum<<<MME, 256, 0, stream>>>(para, mpos, mid, ent_sum, cnt);
    k_build_X<<<NTOT, 256, 0, stream>>>(para, ent_sum, cnt, ppos, X);

    // ----- mention MLP over 50k rows (ent mean + sub + obj share weights) -----
    dim3 blk(256);
    k_gemm_wmma<<<dim3(D / BN, (NTOT + BM - 1) / BM), blk, 0, stream>>>(
        X, mW1b, mb1, H1, NTOT, D, D2, 1, 1);
    k_gemm_wmma<<<dim3(D / BN, (NTOT + BM - 1) / BM), blk, 0, stream>>>(
        H1, mW2b, mb2, F, NTOT, D, D, 0, 0);

    // ----- pair MLP -----
    k_build_pair<<<NPAIR, 256, 0, stream>>>(F, PIN);
    k_gemm_wmma<<<dim3(D / BN, (NPAIR + BM - 1) / BM), blk, 0, stream>>>(
        PIN, pW1b, pb1, HP, NPAIR, D, D4, 1, 1);
    k_gemm_wmma<<<dim3(D / BN, (NPAIR + BM - 1) / BM), blk, 0, stream>>>(
        HP, pW2b, pb2, PE, NPAIR, D, D, 0, 0);

    // ----- batched score MLP: rows = [ent*ctx0 ; pair*ctx1 ; pair*ctx2] -----
    k_build_sin<<<NTOT, 256, 0, stream>>>(F, PE, ctx, SIN);
    k_gemm_wmma<<<dim3(SFEAT / BN, (NTOT + BM - 1) / BM), blk, 0, stream>>>(
        SIN, sW1b, sb1, HS, NTOT, SFEAT, D, 1, 0);
    k_score<<<(NTOT + 7) / 8, 256, 0, stream>>>(HS, sW2, sb2, scores, NTOT);

    // ----- sequential hop steps (small) -----
    k_step0<<<1, 256, 0, stream>>>(scores, probs);
    k_pair_step<<<(NPAIR + 255) / 256, 256, 0, stream>>>(scores + NENT, probs, pso, e1);
    k_norm<<<(NENT + 255) / 256, 256, 0, stream>>>(e1, probs + NENT);
    k_pair_step<<<(NPAIR + 255) / 256, 256, 0, stream>>>(scores + NENT + NPAIR, probs + NENT, pso, e2);
    k_norm<<<(NENT + 255) / 256, 256, 0, stream>>>(e2, probs + 2 * NENT);

    // ----- hop-attention mix -----
    k_final<<<(NENT + 255) / 256, 256, 0, stream>>>(probs, hop, out);
}
